// MultiHeadAttention_11081015623929
// MI455X (gfx1250) — compile-verified, matrix operand order checked by reading, not run on hardware
//
#include <hip/hip_runtime.h>
#include <hip/hip_bf16.h>
#include <math.h>

// Problem constants (match reference)
#define BNUM 4
#define SDIM 2048
#define DDIM 1024
#define HNUM 16
#define DH   64
#define MROWS (BNUM * SDIM)   // 8192

// GEMM blocking
#define BM 128
#define BN 128
#define BK 32
#define LDPITCH 40            // bf16 units per LDS row (32 + 8 pad -> bank-conflict-free)

typedef __attribute__((ext_vector_type(16))) __bf16   v16bf;
typedef __attribute__((ext_vector_type(8)))  float    v8f;
typedef __attribute__((ext_vector_type(4)))  unsigned u32x4;

union FragU { u32x4 q[2]; v16bf v; };

// ---- CDNA5 async global->LDS (guarded; falls back to reg bounce) ----------
#if defined(__gfx1250__) && __has_builtin(__builtin_amdgcn_global_load_async_to_lds_b128) && __has_builtin(__builtin_amdgcn_s_wait_asynccnt)
#define USE_ASYNC_LDS 1
#else
#define USE_ASYNC_LDS 0
#endif

#if USE_ASYNC_LDS
typedef __attribute__((ext_vector_type(4))) int v4i_t;
typedef __attribute__((address_space(1))) v4i_t as1_v4i;   // global int4
typedef __attribute__((address_space(3))) v4i_t as3_v4i;   // LDS int4
__device__ __forceinline__ void async_cp16(const unsigned short* g, unsigned short* l) {
  __builtin_amdgcn_global_load_async_to_lds_b128((as1_v4i*)g, (as3_v4i*)l, 0, 0);
}
__device__ __forceinline__ void async_wait_all() { __builtin_amdgcn_s_wait_asynccnt(0); }
#else
__device__ __forceinline__ void async_cp16(const unsigned short* g, unsigned short* l) {
  *(u32x4*)l = *(const u32x4*)g;
}
__device__ __forceinline__ void async_wait_all() {}
#endif

__device__ __forceinline__ unsigned short f2bf(float f) {
  union { float f; unsigned u; } a;
  a.f = f;
  unsigned u = a.u;
  u += 0x7FFFu + ((u >> 16) & 1u);   // round to nearest even
  return (unsigned short)(u >> 16);
}

// WMMA bf16 fragment from a K-contiguous row-major buffer (global).
// Per ISA 7.12.2: lane L holds row (L%16); elements cover
// K = (e/8)*16 + (L/16)*8 + (e%8)  => two 16B chunks at k0+(L/16)*8, +16.
__device__ __forceinline__ v16bf load_frag_g(const unsigned short* __restrict__ buf,
                                             int row, int rowStride, int k0, int lane) {
  const int g = lane >> 4;
  const unsigned short* p = buf + (size_t)row * rowStride + k0 + g * 8;
  FragU f;
  f.q[0] = *(const u32x4*)(p);
  f.q[1] = *(const u32x4*)(p + 16);
  return f.v;
}

// Same fragment gather, but from an LDS tile with LDPITCH row stride.
__device__ __forceinline__ void lds_frag(const unsigned short* base, int row, int lane,
                                         FragU* out) {
  const int g = lane >> 4;
  const unsigned short* p = base + row * LDPITCH + g * 8;
  out->q[0] = *(const u32x4*)(p);
  out->q[1] = *(const u32x4*)(p + 16);
}

__device__ __forceinline__ v8f wmma_bf16(v16bf a, v16bf b, v8f c) {
  return __builtin_amdgcn_wmma_f32_16x16x32_bf16(
      false, a, false, b, (short)0, c, false, false);
}

// Cooperative stage of a 128x32 bf16 tile (row-major, gpitch) into LDS
// (LDPITCH rows). 512 16B chunks / 256 threads = 2 per thread.
__device__ __forceinline__ void stage_tile(const unsigned short* __restrict__ g,
                                           int gpitch, unsigned short* l, int tid) {
#pragma unroll
  for (int c = 0; c < 2; ++c) {
    const int chunk = tid + c * 256;
    const int row = chunk >> 2;         // 4 chunks per 32-elem row
    const int co  = (chunk & 3) * 8;
    async_cp16(g + (size_t)row * gpitch + co, l + row * LDPITCH + co);
  }
}

// ---------------------------------------------------------------- converts
__global__ void cvt_f32_to_bf16(const float* __restrict__ in,
                                unsigned short* __restrict__ out, int n) {
  int i = blockIdx.x * blockDim.x + threadIdx.x;
  const int stride = gridDim.x * blockDim.x;
  for (; i < n; i += stride) out[i] = f2bf(in[i]);
}

// in: [K, N] f32 row-major  ->  out: [N, K] bf16 row-major (transposed)
__global__ void cvt_transpose(const float* __restrict__ in,
                              unsigned short* __restrict__ out, int K, int N) {
  int i = blockIdx.x * blockDim.x + threadIdx.x;
  const int total = K * N;
  const int stride = gridDim.x * blockDim.x;
  for (; i < total; i += stride) {
    const int k = i / N;
    const int n = i - k * N;
    out[(size_t)n * K + k] = f2bf(in[i]);
  }
}

// ---------------------------------------------------------------- GEMM 1
// qkv = Xb[8192,1024] @ Wqkv + bias. 128x128 block tile, 8 waves,
// double-buffered LDS staging via async-to-LDS. Scatters Q/K [B,H,S,DH] and
// V transposed [B,H,DH,S] in bf16.
__global__ void __launch_bounds__(256)
gemm_qkv(const unsigned short* __restrict__ Xb,
         const unsigned short* __restrict__ Wt,   // [3D, D] bf16 (W^T)
         const float* __restrict__ bias,
         unsigned short* __restrict__ Qm,
         unsigned short* __restrict__ Km,
         unsigned short* __restrict__ Vt) {
  __shared__ unsigned short ldsA[2 * BM * LDPITCH];
  __shared__ unsigned short ldsB[2 * BN * LDPITCH];

  const int NTB = (3 * DDIM) / BN;      // 24 N-tiles
  const int am0 = (blockIdx.x / NTB) * BM;
  const int bn0 = (blockIdx.x % NTB) * BN;

  const int tid  = threadIdx.x;
  const int lane = tid & 31;
  const int wave = tid >> 5;
  const int wm = wave & 3;              // 4 M-groups of 32 rows
  const int wn = wave >> 2;             // 2 N-groups of 64 cols
  const int nl = lane & 15;
  const int g  = lane >> 4;

  v8f acc[2][4] = {};

  stage_tile(Xb + (size_t)am0 * DDIM, DDIM, ldsA, tid);
  stage_tile(Wt + (size_t)bn0 * DDIM, DDIM, ldsB, tid);

  const int NK = DDIM / BK;             // 32
  for (int kt = 0; kt < NK; ++kt) {
    const int cur = kt & 1;
    async_wait_all();                   // my copies of tile kt landed
    __syncthreads();                    // everyone's copies visible; prev compute done
    if (kt + 1 < NK) {                  // prefetch next tile into other buffer
      const int nb = 1 - cur;
      stage_tile(Xb + (size_t)am0 * DDIM + (kt + 1) * BK, DDIM,
                 ldsA + nb * BM * LDPITCH, tid);
      stage_tile(Wt + (size_t)bn0 * DDIM + (kt + 1) * BK, DDIM,
                 ldsB + nb * BN * LDPITCH, tid);
    }
    const unsigned short* At = ldsA + cur * BM * LDPITCH;
    const unsigned short* Bt = ldsB + cur * BN * LDPITCH;
    FragU af[2], bfr[4];
#pragma unroll
    for (int i = 0; i < 2; ++i) lds_frag(At, wm * 32 + i * 16 + nl, lane, &af[i]);
#pragma unroll
    for (int j = 0; j < 4; ++j) lds_frag(Bt, wn * 64 + j * 16 + nl, lane, &bfr[j]);
#pragma unroll
    for (int i = 0; i < 2; ++i)
#pragma unroll
      for (int j = 0; j < 4; ++j)
        acc[i][j] = wmma_bf16(af[i].v, bfr[j].v, acc[i][j]);
  }

  // Epilogue: sect/batch uniform per block (128 | 1024, 128 | 2048).
  const int sect = bn0 / DDIM;                      // 0=Q 1=K 2=V
  const int h    = (bn0 % DDIM) / DH + wn;          // uniform per wave
  const int bIdx = am0 >> 11;
  const size_t bh = (size_t)(bIdx * HNUM + h);
#pragma unroll
  for (int i = 0; i < 2; ++i) {
#pragma unroll
    for (int t = 0; t < 4; ++t) {
      const int n  = bn0 + wn * 64 + t * 16 + nl;
      const int dh = t * 16 + nl;
      const float bv = bias[n];
#pragma unroll
      for (int r = 0; r < 8; ++r) {
        const int m = am0 + wm * 32 + i * 16 + r + 8 * g;
        const int s = m & (SDIM - 1);
        const unsigned short val = f2bf(acc[i][t][r] + bv);
        if (sect == 0)      Qm[(bh * SDIM + s) * DH + dh] = val;
        else if (sect == 1) Km[(bh * SDIM + s) * DH + dh] = val;
        else                Vt[(bh * DH + dh) * SDIM + s] = val;
      }
    }
  }
}

// ---------------------------------------------------------------- attention
// Flash-style: one wave per (b,h,16 query rows). Online softmax over 32-key
// steps; scores/probs never touch HBM. P re-layout C->A via 1KB LDS.
__global__ void __launch_bounds__(32)
attn(const unsigned short* __restrict__ Qm,
     const unsigned short* __restrict__ Km,
     const unsigned short* __restrict__ Vt,
     const int* __restrict__ mask,
     unsigned short* __restrict__ Ctx) {
  __shared__ unsigned short Plds[16 * 32];

  const int lane = threadIdx.x & 31;
  const int qt = blockIdx.x & (SDIM / 16 - 1);   // 0..127
  const int bh = blockIdx.x >> 7;                // 0..63
  const int bIdx = bh >> 4;
  const int h    = bh & 15;
  const int nl = lane & 15;
  const int g  = lane >> 4;

  const unsigned short* Qb = Qm + (size_t)bh * SDIM * DH;
  const unsigned short* Kb = Km + (size_t)bh * SDIM * DH;
  const unsigned short* Vb = Vt + (size_t)bh * DH * SDIM;

  v16bf qf[2];
#pragma unroll
  for (int c = 0; c < 2; ++c)
    qf[c] = load_frag_g(Qb, qt * 16 + nl, DH, c * 32, lane);

  int qrow[8];
#pragma unroll
  for (int r = 0; r < 8; ++r) qrow[r] = qt * 16 + r + 8 * g;

  float mrow[8], lrow[8];
#pragma unroll
  for (int r = 0; r < 8; ++r) { mrow[r] = -1e30f; lrow[r] = 0.f; }
  v8f acc[4] = {};

  for (int j0 = 0; j0 < SDIM; j0 += 32) {
    v8f sc0 = {}, sc1 = {};
#pragma unroll
    for (int c = 0; c < 2; ++c) {
      const v16bf k0f = load_frag_g(Kb, j0 + nl,      DH, c * 32, lane);
      const v16bf k1f = load_frag_g(Kb, j0 + 16 + nl, DH, c * 32, lane);
      sc0 = wmma_bf16(qf[c], k0f, sc0);
      sc1 = wmma_bf16(qf[c], k1f, sc1);
    }

    const int col0 = j0 + nl, col1 = j0 + 16 + nl;
    float p0[8], p1[8], rmax[8];
#pragma unroll
    for (int r = 0; r < 8; ++r) {
      float v0 = sc0[r] * 0.125f;          // 1/sqrt(64)
      float v1 = sc1[r] * 0.125f;
      const int mb = qrow[r] * SDIM;
      v0 = (mask[mb + col0] == 0) ? -1000.f : v0;
      v1 = (mask[mb + col1] == 0) ? -1000.f : v1;
      p0[r] = v0; p1[r] = v1;
      float mx = fmaxf(v0, v1);
      mx = fmaxf(mx, __shfl_xor(mx, 1, 32));
      mx = fmaxf(mx, __shfl_xor(mx, 2, 32));
      mx = fmaxf(mx, __shfl_xor(mx, 4, 32));
      mx = fmaxf(mx, __shfl_xor(mx, 8, 32));
      rmax[r] = mx;
    }

#pragma unroll
    for (int r = 0; r < 8; ++r) {
      const float mnew  = fmaxf(mrow[r], rmax[r]);
      const float alpha = __expf(mrow[r] - mnew);
      const float e0 = __expf(p0[r] - mnew);
      const float e1 = __expf(p1[r] - mnew);
      p0[r] = e0; p1[r] = e1;
      float rs = e0 + e1;
      rs += __shfl_xor(rs, 1, 32);
      rs += __shfl_xor(rs, 2, 32);
      rs += __shfl_xor(rs, 4, 32);
      rs += __shfl_xor(rs, 8, 32);
      lrow[r] = lrow[r] * alpha + rs;
      mrow[r] = mnew;
#pragma unroll
      for (int t = 0; t < 4; ++t) acc[t][r] *= alpha;
    }

    // C-layout probs -> LDS (row-major 16x32 bf16) -> A-fragment.
#pragma unroll
    for (int r = 0; r < 8; ++r) {
      const int lr = r + 8 * g;
      Plds[lr * 32 + nl]      = f2bf(p0[r]);
      Plds[lr * 32 + 16 + nl] = f2bf(p1[r]);
    }
    FragU pf;
    {
      const unsigned short* pp = &Plds[nl * 32 + g * 8];
      pf.q[0] = *(const u32x4*)(pp);
      pf.q[1] = *(const u32x4*)(pp + 16);
    }

#pragma unroll
    for (int t = 0; t < 4; ++t) {
      const v16bf vf = load_frag_g(Vb, t * 16 + nl, SDIM, j0, lane);
      acc[t] = wmma_bf16(pf.v, vf, acc[t]);
    }
  }

#pragma unroll
  for (int r = 0; r < 8; ++r) {
    const float inv = 1.f / lrow[r];
    const int s = qt * 16 + r + 8 * g;
    const size_t base = ((size_t)bIdx * SDIM + s) * DDIM + h * DH;
#pragma unroll
    for (int t = 0; t < 4; ++t)
      Ctx[base + t * 16 + nl] = f2bf(acc[t][r] * inv);
  }
}

// ---------------------------------------------------------------- GEMM 2
// out = Ctx[8192,1024] @ W_out + b_out, f32 output. Same tiled scheme.
__global__ void __launch_bounds__(256)
gemm_out(const unsigned short* __restrict__ Ctx,
         const unsigned short* __restrict__ Wt,   // [D, D] bf16 (W_out^T)
         const float* __restrict__ bias,
         float* __restrict__ Out) {
  __shared__ unsigned short ldsA[2 * BM * LDPITCH];
  __shared__ unsigned short ldsB[2 * BN * LDPITCH];

  const int NTB = DDIM / BN;            // 8
  const int am0 = (blockIdx.x / NTB) * BM;
  const int bn0 = (blockIdx.x % NTB) * BN;

  const int tid  = threadIdx.x;
  const int lane = tid & 31;
  const int wave = tid >> 5;
  const int wm = wave & 3;
  const int wn = wave >> 2;
  const int nl = lane & 15;
  const int g  = lane >> 4;

  v8f acc[2][4] = {};

  stage_tile(Ctx + (size_t)am0 * DDIM, DDIM, ldsA, tid);
  stage_tile(Wt  + (size_t)bn0 * DDIM, DDIM, ldsB, tid);

  const int NK = DDIM / BK;
  for (int kt = 0; kt < NK; ++kt) {
    const int cur = kt & 1;
    async_wait_all();
    __syncthreads();
    if (kt + 1 < NK) {
      const int nb = 1 - cur;
      stage_tile(Ctx + (size_t)am0 * DDIM + (kt + 1) * BK, DDIM,
                 ldsA + nb * BM * LDPITCH, tid);
      stage_tile(Wt  + (size_t)bn0 * DDIM + (kt + 1) * BK, DDIM,
                 ldsB + nb * BN * LDPITCH, tid);
    }
    const unsigned short* At = ldsA + cur * BM * LDPITCH;
    const unsigned short* Bt = ldsB + cur * BN * LDPITCH;
    FragU af[2], bfr[4];
#pragma unroll
    for (int i = 0; i < 2; ++i) lds_frag(At, wm * 32 + i * 16 + nl, lane, &af[i]);
#pragma unroll
    for (int j = 0; j < 4; ++j) lds_frag(Bt, wn * 64 + j * 16 + nl, lane, &bfr[j]);
#pragma unroll
    for (int i = 0; i < 2; ++i)
#pragma unroll
      for (int j = 0; j < 4; ++j)
        acc[i][j] = wmma_bf16(af[i].v, bfr[j].v, acc[i][j]);
  }

#pragma unroll
  for (int i = 0; i < 2; ++i) {
#pragma unroll
    for (int t = 0; t < 4; ++t) {
      const int n = bn0 + wn * 64 + t * 16 + nl;
      const float bv = bias[n];
#pragma unroll
      for (int r = 0; r < 8; ++r) {
        const int m = am0 + wm * 32 + i * 16 + r + 8 * g;
        Out[(size_t)m * DDIM + n] = acc[i][t][r] + bv;   // coalesced
      }
    }
  }
}

// ---------------------------------------------------------------- launcher
extern "C" void kernel_launch(void* const* d_in, const int* in_sizes, int n_in,
                              void* d_out, int out_size, void* d_ws, size_t ws_size,
                              hipStream_t stream) {
  (void)in_sizes; (void)n_in; (void)out_size; (void)ws_size;
  const float* x    = (const float*)d_in[0];
  const float* Wqkv = (const float*)d_in[1];
  const float* bqkv = (const float*)d_in[2];
  const float* Wout = (const float*)d_in[3];
  const float* bout = (const float*)d_in[4];
  const int*   mask = (const int*)d_in[5];
  float* out = (float*)d_out;

  char* ws = (char*)d_ws;
  size_t off = 0;
  auto wsAlloc = [&](size_t bytes) -> char* {
    char* p = ws + off;
    off += (bytes + 255) & ~(size_t)255;
    return p;
  };
  unsigned short* Xb  = (unsigned short*)wsAlloc((size_t)MROWS * DDIM * 2);
  unsigned short* Wqt = (unsigned short*)wsAlloc((size_t)3 * DDIM * DDIM * 2);
  unsigned short* Wot = (unsigned short*)wsAlloc((size_t)DDIM * DDIM * 2);
  unsigned short* Qm  = (unsigned short*)wsAlloc((size_t)MROWS * DDIM * 2);
  unsigned short* Km  = (unsigned short*)wsAlloc((size_t)MROWS * DDIM * 2);
  unsigned short* Vtm = (unsigned short*)wsAlloc((size_t)MROWS * DDIM * 2);
  unsigned short* Ctx = (unsigned short*)wsAlloc((size_t)MROWS * DDIM * 2);

  cvt_f32_to_bf16<<<2048, 256, 0, stream>>>(x, Xb, MROWS * DDIM);
  cvt_transpose<<<2048, 256, 0, stream>>>(Wqkv, Wqt, DDIM, 3 * DDIM);
  cvt_transpose<<<1024, 256, 0, stream>>>(Wout, Wot, DDIM, DDIM);

  // 64 M-tiles x 24 N-tiles
  gemm_qkv<<<(MROWS / BM) * (3 * DDIM / BN), 256, 0, stream>>>(
      Xb, Wqt, bqkv, Qm, Km, Vtm);

  attn<<<BNUM * HNUM * (SDIM / 16), 32, 0, stream>>>(Qm, Km, Vtm, mask, Ctx);

  // 64 x 8 tiles
  gemm_out<<<(MROWS / BM) * (DDIM / BN), 256, 0, stream>>>(
      Ctx, Wot, bout, out);
}
